// HRFormer_Block_59227599012493
// MI455X (gfx1250) — matrix operand order repaired and checked by reading, not compiled, not run on hardware
//
#include <hip/hip_runtime.h>
#include <hip/hip_bf16.h>

typedef __bf16 bf16;
typedef __attribute__((ext_vector_type(8)))  __bf16 v8bf;
typedef __attribute__((ext_vector_type(16))) __bf16 v16bf;
typedef __attribute__((ext_vector_type(8)))  float  v8f;

#define HD __device__ __forceinline__

HD v8f wmma_bf16(v16bf a, v16bf b, v8f c) {
  // D = A(16x32 bf16) * B(32x16 bf16) + C(16x16 f32)
  return __builtin_amdgcn_wmma_f32_16x16x32_bf16(false, a, false, b, (short)0, c, false, false);
}

union V16U { v16bf v; v8bf h[2]; };

// A-matrix fragment (16x32, bf16) from row-major [m][k] with row stride ldr (bf16 elems).
// ISA layout: lane m=L%16; half=L/16; a[0..7] = K(half*8 .. +7); a[8..15] = K(16+half*8 .. +7)
HD v16bf load_a_frag(const bf16* base, int ldr) {
  int lane = threadIdx.x & 31;
  const bf16* p = base + (lane & 15) * ldr + (lane >> 4) * 8;
  V16U u;
  u.h[0] = *(const v8bf*)(p);
  u.h[1] = *(const v8bf*)(p + 16);
  return u.v;
}

// B-matrix fragment (32x16, bf16) from [n][k] storage with row stride ldr.
// ISA layout: lane n=L%16; half=L/16; b[0..7] = K(half*8 .. +7); b[8..15] = K(16+half*8 .. +7)
HD v16bf load_b_frag(const bf16* base, int ldr) {
  int lane = threadIdx.x & 31;
  const bf16* p = base + (lane & 15) * ldr + (lane >> 4) * 8;
  V16U u;
  u.h[0] = *(const v8bf*)(p);
  u.h[1] = *(const v8bf*)(p + 16);
  return u.v;
}

HD v8f v8f_zero() {
  v8f z;
#pragma unroll
  for (int i = 0; i < 8; ++i) z[i] = 0.f;
  return z;
}

HD v8bf v8bf_zero() {
  v8bf z;
#pragma unroll
  for (int i = 0; i < 8; ++i) z[i] = (bf16)0.f;
  return z;
}

HD float leaky(float x)    { return x > 0.f ? x : 0.01f * x; }
HD float softplusf(float x){ return x > 20.f ? x : log1pf(__expf(x)); }

// ---------------------------------------------------------------------------
// Weight packing: fp32 -> bf16, [n][k] (K-contiguous) layouts for WMMA B-frags
// ---------------------------------------------------------------------------
__global__ __launch_bounds__(256)
void pack_weights(const float* __restrict__ W1, const float* __restrict__ W2,
                  const float* __restrict__ W3,
                  bf16* __restrict__ W1p, bf16* __restrict__ W2p, bf16* __restrict__ W3p) {
  const int N1 = 768 * 192;
  const int N2 = 9 * 768 * 768;
  const int N3 = 192 * 768;
  const int total = N1 + N2 + N3;
  for (int i = blockIdx.x * blockDim.x + threadIdx.x; i < total; i += gridDim.x * blockDim.x) {
    if (i < N1) {                       // W1 (768,192,1,1) OIHW -> [co][ci] (identical order)
      W1p[i] = (bf16)W1[i];
    } else if (i < N1 + N2) {           // W2 (768,768,3,3) -> [ky*3+kx][co][ci]
      int d = i - N1;
      int kykx = d / (768 * 768);
      int rem  = d - kykx * (768 * 768);
      int co = rem / 768, ci = rem - co * 768;
      W2p[d] = (bf16)W2[(co * 768 + ci) * 9 + kykx];
    } else {                            // W3 (192,768,1,1) -> [co][ci]
      int d = i - N1 - N2;
      W3p[d] = (bf16)W3[d];
    }
  }
}

// ---------------------------------------------------------------------------
// Attention (softplus linear attention, reassociated r = q @ (k^T v)) + residual
// fused with conv1 (1x1, 192->768) + leaky. One block per 8x8 window of one image.
// Writes h1 as bf16 NHWC (256,256,768).
// ---------------------------------------------------------------------------
#define LDA1 208   // padded row (bf16 elems) for the 64x192 A-tile (16B aligned rows)

__global__ __launch_bounds__(256)
void attn_conv1_kernel(const float* __restrict__ x,   // one image (192,256,256)
                       const float* __restrict__ Wq, const float* __restrict__ bq,
                       const float* __restrict__ Wk, const float* __restrict__ bk,
                       const float* __restrict__ Wv, const float* __restrict__ bv,
                       const float* __restrict__ Wo, const float* __restrict__ bo,
                       const bf16*  __restrict__ W1p, const float* __restrict__ b1,
                       bf16* __restrict__ h1) {
  // ubuf aliases: phase 1 = fp32 window half (96 rows x 65), phase 2 = bf16 A-tile 64x208
  __shared__ alignas(16) char ubuf[64 * LDA1 * 2];     // 26,624 B
  __shared__ float qs[192 * 8];
  __shared__ float ks[192 * 8];                        // later reused to hold r
  __shared__ float vs[192 * 8];
  __shared__ float kvb[64];
  __shared__ float wqs[512], wks[512], wvs[512], wos[512];
  __shared__ float bqs[8], bks[8], bvs[8], bos[64];

  float* xs = (float*)ubuf;       // [96][65]
  bf16*  As = (bf16*)ubuf;        // [64][LDA1]

  const int t = threadIdx.x;
  const int win = blockIdx.x;
  const int y0 = (win >> 5) * 8;
  const int x0 = (win & 31) * 8;

  for (int idx = t; idx < 512; idx += 256) {
    wqs[idx] = Wq[idx]; wks[idx] = Wk[idx]; wvs[idx] = Wv[idx]; wos[idx] = Wo[idx];
  }
  if (t < 8)  { bqs[t] = bq[t]; bks[t] = bk[t]; bvs[t] = bv[t]; }
  if (t < 64) bos[t] = bo[t];

  // q,k,v projections over p (64 pixels), in two channel halves to bound LDS
  for (int hh = 0; hh < 2; ++hh) {
    const int cbase = hh * 96;
    __syncthreads();
    for (int idx = t; idx < 96 * 64; idx += 256) {
      int c = idx >> 6, p = idx & 63;
      xs[c * 65 + p] = x[((cbase + c) * 256 + y0 + (p >> 3)) * 256 + x0 + (p & 7)];
    }
    __syncthreads();
    for (int idx = t; idx < 96 * 8; idx += 256) {
      int c = idx >> 3, d = idx & 7;
      const float* xr = &xs[c * 65];
      float sq = bqs[d], sk = bks[d], sv = bvs[d];
#pragma unroll 8
      for (int p = 0; p < 64; ++p) {
        float xv = xr[p];
        sq += xv * wqs[p * 8 + d];
        sk += xv * wks[p * 8 + d];
        sv += xv * wvs[p * 8 + d];
      }
      int cg = cbase + c;
      qs[cg * 8 + d] = softplusf(sq);
      ks[cg * 8 + d] = softplusf(sk);
      vs[cg * 8 + d] = sv;
    }
  }
  __syncthreads();

  // kv[e][d] = sum_c k[c][e] * v[c][d]   (8x8)
  if (t < 64) {
    int e = t >> 3, d = t & 7;
    float s = 0.f;
    for (int c = 0; c < 192; ++c) s += ks[c * 8 + e] * vs[c * 8 + d];
    kvb[t] = s;
  }
  __syncthreads();

  // r = q @ kv -> overwrite ks
  for (int idx = t; idx < 1536; idx += 256) {
    int c = idx >> 3, d = idx & 7;
    float s = 0.f;
#pragma unroll
    for (int e = 0; e < 8; ++e) s += qs[c * 8 + e] * kvb[e * 8 + d];
    ks[c * 8 + d] = s;
  }
  __syncthreads();

  // y = x + r@Wo + bo  ->  bf16 A-tile As[p][c]  (residual re-reads x from global/L2)
  for (int idx = t; idx < 12288; idx += 256) {
    int c = idx >> 6, p = idx & 63;
    const float* rr = &ks[c * 8];
    float s = x[(c * 256 + y0 + (p >> 3)) * 256 + x0 + (p & 7)] + bos[p];
#pragma unroll
    for (int d = 0; d < 8; ++d) s += rr[d] * wos[d * 64 + p];
    As[p * LDA1 + c] = (bf16)s;
  }
  __syncthreads();

  // conv1 GEMM: (64 x 192) @ (192 x 768), B-frags straight from L2-resident W1p
  const int w = t >> 5, lane = t & 31;
  const int wave_m = w >> 2;            // 0..1 -> m base 0/32
  const int wave_n = w & 3;             // 0..3 -> n base *192
  const int mb = wave_m * 32;
  const int half = lane >> 4, nl = lane & 15;

  for (int ng = 0; ng < 2; ++ng) {
    const int nb = wave_n * 192 + ng * 96;
    v8f acc[2][6];
#pragma unroll
    for (int mf = 0; mf < 2; ++mf)
#pragma unroll
      for (int nf = 0; nf < 6; ++nf) acc[mf][nf] = v8f_zero();

    for (int k0 = 0; k0 < 192; k0 += 32) {
      v16bf a0 = load_a_frag(&As[(mb +  0) * LDA1 + k0], LDA1);
      v16bf a1 = load_a_frag(&As[(mb + 16) * LDA1 + k0], LDA1);
#pragma unroll
      for (int nf = 0; nf < 6; ++nf) {
        v16bf b = load_b_frag(W1p + (nb + nf * 16) * 192 + k0, 192);
        acc[0][nf] = wmma_bf16(a0, b, acc[0][nf]);
        acc[1][nf] = wmma_bf16(a1, b, acc[1][nf]);
      }
    }
#pragma unroll
    for (int mf = 0; mf < 2; ++mf) {
#pragma unroll
      for (int nf = 0; nf < 6; ++nf) {
        int co = nb + nf * 16 + nl;
        float bias = b1[co];
#pragma unroll
        for (int i = 0; i < 8; ++i) {
          int p = mb + mf * 16 + i + 8 * half;
          int yy = y0 + (p >> 3), xx = x0 + (p & 7);
          h1[(yy * 256 + xx) * 768 + co] = (bf16)leaky(acc[mf][nf][i] + bias);
        }
      }
    }
  }
}

// ---------------------------------------------------------------------------
// conv2: 3x3 SAME, 768->768, implicit GEMM. Block tile: 128 pixels x 256 couts.
// A tile (130 px halo x 32 cin) reused across the 3 kx taps; B staged via LDS.
// ---------------------------------------------------------------------------
#define LDT 40     // padded K-row (bf16 elems): 80B rows, conflict-free b128 frag reads

__global__ __launch_bounds__(256)
void conv2_kernel(const bf16* __restrict__ h1, const bf16* __restrict__ W2p,
                  const float* __restrict__ b2, bf16* __restrict__ h2) {
  __shared__ alignas(16) bf16 Atile[130 * LDT];   // 10,400 B
  __shared__ alignas(16) bf16 Btile[256 * LDT];   // 20,480 B

  const int t = threadIdx.x;
  const int x0 = blockIdx.x * 128;
  const int y  = blockIdx.y;
  const int n0 = blockIdx.z * 256;
  const int w = t >> 5, lane = t & 31;
  const int wave_m = w & 1, wave_n = w >> 1;      // 2 x 4 waves, each 64x64
  const int half = lane >> 4, nl = lane & 15;

  v8f acc[4][4];
#pragma unroll
  for (int mf = 0; mf < 4; ++mf)
#pragma unroll
    for (int nf = 0; nf < 4; ++nf) acc[mf][nf] = v8f_zero();

  for (int ky = 0; ky < 3; ++ky) {
    int ysrc = y + ky - 1;
    if (ysrc < 0 || ysrc >= 256) continue;        // zero padding: no contribution
    const bf16* arow = h1 + (size_t)ysrc * 256 * 768;
    for (int c0 = 0; c0 < 768; c0 += 32) {
      __syncthreads();                            // protect Atile from prior reads
      for (int idx = t; idx < 520; idx += 256) {  // 130 px * 4 chunks of 8 bf16
        int pix = idx >> 2, part = idx & 3;
        int xsrc = x0 - 1 + pix;
        v8bf v = (xsrc >= 0 && xsrc < 256)
                   ? *(const v8bf*)(arow + (size_t)xsrc * 768 + c0 + part * 8)
                   : v8bf_zero();
        *(v8bf*)(&Atile[pix * LDT + part * 8]) = v;
      }
      for (int kx = 0; kx < 3; ++kx) {
        __syncthreads();                          // protect Btile / publish Atile
        for (int idx = t; idx < 1024; idx += 256) {
          int col = idx >> 2, part = idx & 3;
          const bf16* src = W2p + ((size_t)(ky * 3 + kx) * 768 + n0 + col) * 768 + c0 + part * 8;
          *(v8bf*)(&Btile[col * LDT + part * 8]) = *(const v8bf*)(src);
        }
        __syncthreads();
        v16bf a[4];
#pragma unroll
        for (int mf = 0; mf < 4; ++mf)
          a[mf] = load_a_frag(&Atile[(wave_m * 64 + mf * 16 + kx) * LDT], LDT);
#pragma unroll
        for (int nf = 0; nf < 4; ++nf) {
          v16bf b = load_b_frag(&Btile[(wave_n * 64 + nf * 16) * LDT], LDT);
#pragma unroll
          for (int mf = 0; mf < 4; ++mf)
            acc[mf][nf] = wmma_bf16(a[mf], b, acc[mf][nf]);
        }
      }
    }
  }

#pragma unroll
  for (int mf = 0; mf < 4; ++mf) {
#pragma unroll
    for (int nf = 0; nf < 4; ++nf) {
      int co = n0 + wave_n * 64 + nf * 16 + nl;
      float bias = b2[co];
#pragma unroll
      for (int i = 0; i < 8; ++i) {
        int m = wave_m * 64 + mf * 16 + i + 8 * half;
        h2[((size_t)y * 256 + x0 + m) * 768 + co] = (bf16)leaky(acc[mf][nf][i] + bias);
      }
    }
  }
}

// ---------------------------------------------------------------------------
// conv3: 1x1, 768->192, + leaky. Block tile: 128 pixels x 192 couts (all couts).
// Epilogue transposes via LDS so NCHW fp32 stores are contiguous along x.
// ---------------------------------------------------------------------------
__global__ __launch_bounds__(256)
void conv3_kernel(const bf16* __restrict__ h2, const bf16* __restrict__ W3p,
                  const float* __restrict__ b3, float* __restrict__ out) {
  __shared__ alignas(16) char buf3[64 * 193 * 4];   // 49,408 B, aliased
  bf16*  As3   = (bf16*)buf3;                       // [128][LDT]  = 10,240 B
  bf16*  Bs3   = (bf16*)(buf3 + 10240);             // [192][LDT]  = 15,360 B
  float* stage = (float*)buf3;                      // [64][193] fp32 in epilogue

  const int t = threadIdx.x;
  const int x0 = blockIdx.x * 128;
  const int y  = blockIdx.y;
  const int w = t >> 5, lane = t & 31;
  const int wave_m = w & 1, wave_n = w >> 1;        // 2 x 4 waves, each 64x48
  const int half = lane >> 4, nl = lane & 15;

  v8f acc[4][3];
#pragma unroll
  for (int mf = 0; mf < 4; ++mf)
#pragma unroll
    for (int nf = 0; nf < 3; ++nf) acc[mf][nf] = v8f_zero();

  const bf16* arow = h2 + (size_t)y * 256 * 768;
  for (int c0 = 0; c0 < 768; c0 += 32) {
    __syncthreads();
    for (int idx = t; idx < 512; idx += 256) {      // 128 px * 4 chunks
      int pix = idx >> 2, part = idx & 3;
      *(v8bf*)(&As3[pix * LDT + part * 8]) =
          *(const v8bf*)(arow + (size_t)(x0 + pix) * 768 + c0 + part * 8);
    }
    for (int idx = t; idx < 768; idx += 256) {      // 192 co * 4 chunks
      int col = idx >> 2, part = idx & 3;
      *(v8bf*)(&Bs3[col * LDT + part * 8]) =
          *(const v8bf*)(W3p + (size_t)col * 768 + c0 + part * 8);
    }
    __syncthreads();
    v16bf a[4];
#pragma unroll
    for (int mf = 0; mf < 4; ++mf)
      a[mf] = load_a_frag(&As3[(wave_m * 64 + mf * 16) * LDT], LDT);
#pragma unroll
    for (int nf = 0; nf < 3; ++nf) {
      v16bf b = load_b_frag(&Bs3[(wave_n * 48 + nf * 16) * LDT], LDT);
#pragma unroll
      for (int mf = 0; mf < 4; ++mf)
        acc[mf][nf] = wmma_bf16(a[mf], b, acc[mf][nf]);
    }
  }

  // Epilogue in two 64-pixel rounds (wave_m 0 then 1), transposed through LDS.
  for (int r = 0; r < 2; ++r) {
    __syncthreads();
    if (wave_m == r) {
#pragma unroll
      for (int mf = 0; mf < 4; ++mf) {
#pragma unroll
        for (int nf = 0; nf < 3; ++nf) {
          int n = wave_n * 48 + nf * 16 + nl;
          float bias = b3[n];
#pragma unroll
          for (int i = 0; i < 8; ++i) {
            int m = mf * 16 + i + 8 * half;          // 0..63 within this half
            stage[m * 193 + n] = leaky(acc[mf][nf][i] + bias);
          }
        }
      }
    }
    __syncthreads();
    for (int idx = t; idx < 64 * 192; idx += 256) {
      int n = idx >> 6, m = idx & 63;
      out[((size_t)n * 256 + y) * 256 + x0 + r * 64 + m] = stage[m * 193 + n];
    }
  }
}

// ---------------------------------------------------------------------------
extern "C" void kernel_launch(void* const* d_in, const int* in_sizes, int n_in,
                              void* d_out, int out_size, void* d_ws, size_t ws_size,
                              hipStream_t stream) {
  const float* x  = (const float*)d_in[0];
  const float* Wq = (const float*)d_in[1];
  const float* bq = (const float*)d_in[2];
  const float* Wk = (const float*)d_in[3];
  const float* bk = (const float*)d_in[4];
  const float* Wv = (const float*)d_in[5];
  const float* bv = (const float*)d_in[6];
  const float* Wo = (const float*)d_in[7];
  const float* bo = (const float*)d_in[8];
  const float* W1 = (const float*)d_in[9];
  const float* b1 = (const float*)d_in[10];
  const float* W2 = (const float*)d_in[11];
  const float* b2 = (const float*)d_in[12];
  const float* W3 = (const float*)d_in[13];
  const float* b3 = (const float*)d_in[14];
  float* out = (float*)d_out;

  char* ws = (char*)d_ws;
  size_t off = 0;
  auto take = [&](size_t bytes) { void* p = ws + off; off = (off + bytes + 255) & ~(size_t)255; return p; };
  bf16* W1p = (bf16*)take((size_t)768 * 192 * 2);
  bf16* W2p = (bf16*)take((size_t)9 * 768 * 768 * 2);
  bf16* W3p = (bf16*)take((size_t)192 * 768 * 2);
  bf16* h1  = (bf16*)take((size_t)256 * 256 * 768 * 2);
  bf16* h2  = (bf16*)take((size_t)256 * 256 * 768 * 2);

  pack_weights<<<4096, 256, 0, stream>>>(W1, W2, W3, W1p, W2p, W3p);

  for (int b = 0; b < 4; ++b) {   // per-image: keeps h1/h2 (96 MB each) L2-resident
    const float* xb = x + (size_t)b * 192 * 256 * 256;
    float* outb = out + (size_t)b * 192 * 256 * 256;
    attn_conv1_kernel<<<1024, 256, 0, stream>>>(xb, Wq, bq, Wk, bk, Wv, bv, Wo, bo,
                                                W1p, b1, h1);
    conv2_kernel<<<dim3(2, 256, 3), 256, 0, stream>>>(h1, W2p, b2, h2);
    conv3_kernel<<<dim3(2, 256, 1), 256, 0, stream>>>(h2, W3p, b3, outb);
  }
}
